// MobileMQA1D_79156247265882
// MI455X (gfx1250) — matrix operand order repaired and checked by reading, not compiled, not run on hardware
//
#include <hip/hip_runtime.h>
#include <hip/hip_bf16.h>

#define B_ 4
#define C_ 512
#define L_ 2048
#define H_ 8
#define D_ 64

typedef __attribute__((ext_vector_type(16))) __bf16 v16bf;
typedef __attribute__((ext_vector_type(8)))  float  v8f;
typedef __attribute__((ext_vector_type(4)))  unsigned int u32x4;
typedef __attribute__((ext_vector_type(8)))  int i32x8;
typedef __attribute__((ext_vector_type(4)))  int i32x4;

static __device__ __forceinline__ v8f wmma_bf16(v16bf a, v16bf b, v8f c) {
  return __builtin_amdgcn_wmma_f32_16x16x32_bf16(false, a, false, b, (short)0, c,
                                                 false, false);
}

// ---- A operand fragment: 16x32 (MxK), element (m,k) at src[m*ld + k] ----
static __device__ __forceinline__ v16bf load_a_frag(const __bf16* src, int ld) {
  const int lane = threadIdx.x & 31;
  const int m = lane & 15;
  const int kb = (lane >> 4) << 4; // 0 or 16
  const unsigned int* p = reinterpret_cast<const unsigned int*>(src + (size_t)m * ld);
  union { v16bf v; unsigned int d[8]; } u;
#pragma unroll
  for (int j = 0; j < 4; ++j) u.d[j] = p[(kb >> 1) + j];          // K = kb+2j
#pragma unroll
  for (int j = 0; j < 4; ++j) u.d[4 + j] = p[4 + (kb >> 1) + j];  // K = 8+kb+2j
  return u.v;
}

// ---- B operand fragment: 32x16 (KxN), element (k,n) at src[n*ld + k] ----
static __device__ __forceinline__ v16bf load_b_frag(const __bf16* src, int ld) {
  const int lane = threadIdx.x & 31;
  const int n = lane & 15;
  const int ks = (lane >> 4) << 4; // 0 or 16
  const unsigned int* p =
      reinterpret_cast<const unsigned int*>(src + (size_t)n * ld + ks);
  union { v16bf v; unsigned int d[8]; } u;
#pragma unroll
  for (int j = 0; j < 8; ++j) u.d[j] = p[j]; // 32B contiguous
  return u.v;
}

static __device__ __forceinline__ v8f vzero8() {
  v8f z = {0.f, 0.f, 0.f, 0.f, 0.f, 0.f, 0.f, 0.f};
  return z;
}

// ---- Tensor Data Mover: 2-D tile (16-bit elements) global -> LDS ----
// D# packing per CDNA5 ISA; 6-arg clang-23 builtin.
static __device__ __forceinline__ void tdm_load_2d(unsigned int lds_byte_addr,
                                                   const void* gptr,
                                                   unsigned int tensor_d0,
                                                   unsigned int tensor_d1,
                                                   unsigned int tile_d0,
                                                   unsigned int tile_d1,
                                                   unsigned long long stride0) {
  unsigned long long ga = (unsigned long long)(size_t)gptr;
  union { u32x4 v; unsigned int a[4]; } g0;
  g0.a[0] = 1u;                                    // count=1 (valid user D#)
  g0.a[1] = lds_byte_addr;                         // lds_addr [63:32]
  g0.a[2] = (unsigned int)ga;                      // global_addr [95:64]
  g0.a[3] = (unsigned int)(ga >> 32) | (2u << 30); // addr hi [120:96] | type=2
  union { i32x8 v; unsigned int a[8]; } g1;
  g1.a[0] = (1u << 16);                            // wg_mask=0, data_size=1 (2B)
  g1.a[1] = (tensor_d0 & 0xFFFFu) << 16;           // tensor_dim0[15:0]
  g1.a[2] = (tensor_d0 >> 16) | ((tensor_d1 & 0xFFFFu) << 16);
  g1.a[3] = (tensor_d1 >> 16) | (tile_d0 << 16);   // tile_dim0 [127:112]
  g1.a[4] = tile_d1;                               // tile_dim1, tile_dim2=0
  g1.a[5] = (unsigned int)stride0;                 // tensor_dim0_stride lo
  g1.a[6] = (unsigned int)(stride0 >> 32);         // stride hi16, dim1_stride=0
  g1.a[7] = 0;
  i32x4 z4 = {0, 0, 0, 0};
  i32x8 z8 = {0, 0, 0, 0, 0, 0, 0, 0};
  __builtin_amdgcn_tensor_load_to_lds(g0.v, g1.v, z4, z4, z8, 0);
}

static __device__ __forceinline__ unsigned int lds_addr_of(const void* p) {
  return (unsigned int)(size_t)p; // generic LDS ptr: addr[31:0] == LDS offset
}

// ---------------- pack kernels ----------------
__global__ void pack_x_kernel(const float* __restrict__ x, __bf16* __restrict__ xp) {
  int idx = blockIdx.x * blockDim.x + threadIdx.x;
  if (idx >= B_ * C_ * L_) return;
  int l = idx % L_;
  int c = (idx / L_) % C_;
  int b = idx / (C_ * L_);
  xp[((size_t)b * L_ + l) * C_ + c] = (__bf16)x[idx]; // (B,C,L) -> (B,L,C) bf16
}

__global__ void pack_w_kernel(const float* __restrict__ w, __bf16* __restrict__ wp,
                              int n) {
  int idx = blockIdx.x * blockDim.x + threadIdx.x;
  if (idx < n) wp[idx] = (__bf16)w[idx];
}

// ---------------- QKV projection ----------------
__global__ void qkv_kernel(const __bf16* __restrict__ xp,
                           const __bf16* __restrict__ Wq, const __bf16* __restrict__ Wk,
                           const __bf16* __restrict__ Wv,
                           const float* __restrict__ bq, const float* __restrict__ bk,
                           const float* __restrict__ bv,
                           __bf16* __restrict__ q, __bf16* __restrict__ k,
                           __bf16* __restrict__ vT) {
  const int wave = threadIdx.x >> 5;
  const int task = blockIdx.x * (blockDim.x >> 5) + wave;
  const int per_mat = B_ * (L_ / 16) * (C_ / 64); // 4096
  if (task >= 3 * per_mat) return;
  const int mat = task / per_mat;
  const int r = task % per_mat;
  const int ng = r % (C_ / 64);
  const int lt = (r / (C_ / 64)) % (L_ / 16);
  const int b = r / ((C_ / 64) * (L_ / 16));

  const __bf16* W = (mat == 0) ? Wq : (mat == 1) ? Wk : Wv;
  const float* bias = (mat == 0) ? bq : (mat == 1) ? bk : bv;

  const __bf16* A = xp + ((size_t)b * L_ + lt * 16) * C_;
  const __bf16* Bw = W + (size_t)(ng * 64) * C_;

  v8f acc[4] = {vzero8(), vzero8(), vzero8(), vzero8()};
#pragma unroll 1
  for (int kk = 0; kk < C_ / 32; ++kk) {
    v16bf a = load_a_frag(A + kk * 32, C_);
#pragma unroll
    for (int nt = 0; nt < 4; ++nt) {
      v16bf bb = load_b_frag(Bw + (size_t)(nt * 16) * C_ + kk * 32, C_);
      acc[nt] = wmma_bf16(a, bb, acc[nt]);
    }
  }

  const int lane = threadIdx.x & 31;
  const int ncol = lane & 15;
  const int hf = lane >> 4;
#pragma unroll
  for (int nt = 0; nt < 4; ++nt) {
    const int d = nt * 16 + ncol;
    const float bval = bias[ng * 64 + d];
    if (mat < 2) {
      __bf16* dst = (mat == 0) ? q : k;
      size_t base = (((size_t)b * H_ + ng) * L_ + lt * 16 + hf * 8) * D_ + d;
#pragma unroll
      for (int i = 0; i < 8; ++i)
        dst[base + (size_t)i * D_] = (__bf16)(acc[nt][i] + bval);
    } else {
      size_t base = (((size_t)b * H_ + ng) * D_ + d) * L_ + lt * 16 + hf * 8;
#pragma unroll
      for (int i = 0; i < 8; ++i)
        vT[base + i] = (__bf16)(acc[nt][i] + bval);
    }
  }
}

// ---------------- attention (block-cooperative, TDM-staged) ----------------
// 1 block = 8 waves = one (b,h) x 128 queries. 64-key blocks double-buffered in
// LDS by the Tensor Data Mover (wave 0 issues, s_wait_tensorcnt + barrier).
__global__ void __launch_bounds__(256) attn_kernel(const __bf16* __restrict__ q,
                                                   const __bf16* __restrict__ k,
                                                   const __bf16* __restrict__ vT,
                                                   __bf16* __restrict__ o) {
  __shared__ __bf16 kbuf[2][64 * D_];   // 2 x 8 KB, (key, d) ld=64
  __shared__ __bf16 vbuf[2][D_ * 64];   // 2 x 8 KB, (d, key) ld=64
  __shared__ __bf16 ptile[8][16 * 64];  // 2 KB per wave

  const int wave = threadIdx.x >> 5;
  const int lane = threadIdx.x & 31;
  const int ncol = lane & 15;
  const int hf = lane >> 4;
  const int bh = blockIdx.x >> 4;  // b*H + h
  const int qg = blockIdx.x & 15;  // group of 128 queries
  const int qt = qg * 8 + wave;    // this wave's 16-query tile
  const float scale = 0.125f;      // D^-0.5
  const int NB = L_ / 64;          // 32 key blocks

  const __bf16* qbase = q + ((size_t)bh * L_ + qt * 16) * D_;
  const v16bf qa0 = load_a_frag(qbase, D_);
  const v16bf qa1 = load_a_frag(qbase + 32, D_);

  const __bf16* kh = k + (size_t)bh * L_ * D_;
  const __bf16* vh = vT + (size_t)bh * D_ * L_;

  // ---- pass 1: row max of scaled scores ----
  float mrow[8];
#pragma unroll
  for (int i = 0; i < 8; ++i) mrow[i] = -3.0e38f;

  if (wave == 0)
    tdm_load_2d(lds_addr_of(&kbuf[0][0]), kh, D_, L_, 64, 64, D_);
#pragma unroll 1
  for (int kt = 0; kt < NB; ++kt) {
    if (wave == 0) __builtin_amdgcn_s_wait_tensorcnt(0);
    __syncthreads();
    if (wave == 0 && kt + 1 < NB)
      tdm_load_2d(lds_addr_of(&kbuf[(kt + 1) & 1][0]),
                  kh + (size_t)(kt + 1) * 64 * D_, D_, L_, 64, 64, D_);
    const __bf16* kb = &kbuf[kt & 1][0];
#pragma unroll
    for (int sub = 0; sub < 4; ++sub) {
      v16bf b0 = load_b_frag(kb + sub * 16 * D_, D_);
      v16bf b1 = load_b_frag(kb + sub * 16 * D_ + 32, D_);
      v8f s = vzero8();
      s = wmma_bf16(qa0, b0, s);
      s = wmma_bf16(qa1, b1, s);
#pragma unroll
      for (int i = 0; i < 8; ++i) mrow[i] = fmaxf(mrow[i], s[i] * scale);
    }
  }
  // cross-lane max within each 16-lane half
#pragma unroll
  for (int m = 1; m < 16; m <<= 1)
#pragma unroll
    for (int i = 0; i < 8; ++i)
      mrow[i] = fmaxf(mrow[i], __shfl_xor(mrow[i], m, 16));

  // ---- pass 2: exp / rowsum / P@V ----
  v8f acc[4] = {vzero8(), vzero8(), vzero8(), vzero8()};
  float rsum[8] = {0.f, 0.f, 0.f, 0.f, 0.f, 0.f, 0.f, 0.f};
  __bf16* pt = &ptile[wave][0];

  if (wave == 0) {
    tdm_load_2d(lds_addr_of(&kbuf[0][0]), kh, D_, L_, 64, 64, D_);
    tdm_load_2d(lds_addr_of(&vbuf[0][0]), vh, L_, D_, 64, 64, L_);
  }
#pragma unroll 1
  for (int kt = 0; kt < NB; ++kt) {
    if (wave == 0) __builtin_amdgcn_s_wait_tensorcnt(0);
    __syncthreads();
    if (wave == 0 && kt + 1 < NB) {
      tdm_load_2d(lds_addr_of(&kbuf[(kt + 1) & 1][0]),
                  kh + (size_t)(kt + 1) * 64 * D_, D_, L_, 64, 64, D_);
      tdm_load_2d(lds_addr_of(&vbuf[(kt + 1) & 1][0]),
                  vh + (size_t)(kt + 1) * 64, L_, D_, 64, 64, L_);
    }
    const __bf16* kb = &kbuf[kt & 1][0];
    const __bf16* vbb = &vbuf[kt & 1][0];
#pragma unroll
    for (int sub = 0; sub < 4; ++sub) {
      v16bf b0 = load_b_frag(kb + sub * 16 * D_, D_);
      v16bf b1 = load_b_frag(kb + sub * 16 * D_ + 32, D_);
      v8f s = vzero8();
      s = wmma_bf16(qa0, b0, s);
      s = wmma_bf16(qa1, b1, s);
      const int col = sub * 16 + ncol;
#pragma unroll
      for (int i = 0; i < 8; ++i) {
        float p = __expf(s[i] * scale - mrow[i]);
        rsum[i] += p;
        pt[(i + hf * 8) * 64 + col] = (__bf16)p; // C-layout -> per-wave LDS tile
      }
    }
    asm volatile("" ::: "memory");
    v16bf pa0 = load_a_frag(pt, 64);       // probs, keys 0..31
    v16bf pa1 = load_a_frag(pt + 32, 64);  // probs, keys 32..63
#pragma unroll
    for (int nt = 0; nt < 4; ++nt) {
      v16bf vb0 = load_b_frag(vbb + (size_t)(nt * 16) * 64, 64);
      v16bf vb1 = load_b_frag(vbb + (size_t)(nt * 16) * 64 + 32, 64);
      acc[nt] = wmma_bf16(pa0, vb0, acc[nt]);
      acc[nt] = wmma_bf16(pa1, vb1, acc[nt]);
    }
    asm volatile("" ::: "memory");
  }
  // row sums within each 16-lane half
#pragma unroll
  for (int m = 1; m < 16; m <<= 1)
#pragma unroll
    for (int i = 0; i < 8; ++i) rsum[i] += __shfl_xor(rsum[i], m, 16);

  // normalize + store o as (B, L, C) bf16
  const int b = bh / H_;
  const int hh = bh % H_;
#pragma unroll
  for (int nt = 0; nt < 4; ++nt) {
#pragma unroll
    for (int i = 0; i < 8; ++i) {
      const int l = qt * 16 + hf * 8 + i;
      const int c = hh * 64 + nt * 16 + ncol;
      o[((size_t)b * L_ + l) * C_ + c] = (__bf16)(acc[nt][i] / rsum[i]);
    }
  }
}

// ---------------- output projection + residual ----------------
__global__ void proj_kernel(const __bf16* __restrict__ o, const __bf16* __restrict__ Wo,
                            const float* __restrict__ bo, const float* __restrict__ x,
                            float* __restrict__ out) {
  const int wave = threadIdx.x >> 5;
  const int task = blockIdx.x * (blockDim.x >> 5) + wave;
  if (task >= B_ * (L_ / 16) * (C_ / 64)) return;
  const int ng = task % (C_ / 64);
  const int lt = (task / (C_ / 64)) % (L_ / 16);
  const int b = task / ((C_ / 64) * (L_ / 16));

  const __bf16* A = o + ((size_t)b * L_ + lt * 16) * C_;
  const __bf16* Bw = Wo + (size_t)(ng * 64) * C_;

  v8f acc[4] = {vzero8(), vzero8(), vzero8(), vzero8()};
#pragma unroll 1
  for (int kk = 0; kk < C_ / 32; ++kk) {
    v16bf a = load_a_frag(A + kk * 32, C_);
#pragma unroll
    for (int nt = 0; nt < 4; ++nt) {
      v16bf bb = load_b_frag(Bw + (size_t)(nt * 16) * C_ + kk * 32, C_);
      acc[nt] = wmma_bf16(a, bb, acc[nt]);
    }
  }

  const int lane = threadIdx.x & 31;
  const int ncol = lane & 15;
  const int hf = lane >> 4;
#pragma unroll
  for (int nt = 0; nt < 4; ++nt) {
    const int c = ng * 64 + nt * 16 + ncol;
    const float bval = bo[c];
    size_t base = ((size_t)b * C_ + c) * L_ + lt * 16 + hf * 8;
#pragma unroll
    for (int i = 0; i < 8; ++i)
      out[base + i] = x[base + i] + acc[nt][i] + bval;
  }
}

// ---------------- launcher ----------------
extern "C" void kernel_launch(void* const* d_in, const int* in_sizes, int n_in,
                              void* d_out, int out_size, void* d_ws, size_t ws_size,
                              hipStream_t stream) {
  const float* x = (const float*)d_in[0];
  const float* Wq = (const float*)d_in[1];
  const float* bq = (const float*)d_in[2];
  const float* Wk = (const float*)d_in[3];
  const float* bk = (const float*)d_in[4];
  const float* Wv = (const float*)d_in[5];
  const float* bv = (const float*)d_in[6];
  const float* Wo = (const float*)d_in[7];
  const float* bo = (const float*)d_in[8];
  float* out = (float*)d_out;

  char* ws = (char*)d_ws;
  size_t off = 0;
  const size_t xpBytes = (size_t)B_ * L_ * C_ * sizeof(__bf16); // 8 MB
  const size_t wBytes = (size_t)C_ * C_ * sizeof(__bf16);       // 512 KB
  __bf16* xp = (__bf16*)(ws + off); off += xpBytes;
  __bf16* wqb = (__bf16*)(ws + off); off += wBytes;
  __bf16* wkb = (__bf16*)(ws + off); off += wBytes;
  __bf16* wvb = (__bf16*)(ws + off); off += wBytes;
  __bf16* wob = (__bf16*)(ws + off); off += wBytes;
  __bf16* qbuf = (__bf16*)(ws + off); off += xpBytes; // (B,H,L,D)
  __bf16* kbuf = (__bf16*)(ws + off); off += xpBytes; // (B,H,L,D)
  __bf16* vtb = (__bf16*)(ws + off); off += xpBytes;  // (B,H,D,L)
  __bf16* obuf = (__bf16*)(ws + off); off += xpBytes; // (B,L,C)
  (void)off; // ~42 MB required

  pack_x_kernel<<<(B_ * C_ * L_ + 255) / 256, 256, 0, stream>>>(x, xp);
  pack_w_kernel<<<(C_ * C_ + 255) / 256, 256, 0, stream>>>(Wq, wqb, C_ * C_);
  pack_w_kernel<<<(C_ * C_ + 255) / 256, 256, 0, stream>>>(Wk, wkb, C_ * C_);
  pack_w_kernel<<<(C_ * C_ + 255) / 256, 256, 0, stream>>>(Wv, wvb, C_ * C_);
  pack_w_kernel<<<(C_ * C_ + 255) / 256, 256, 0, stream>>>(Wo, wob, C_ * C_);

  qkv_kernel<<<12288 / 8, 256, 0, stream>>>(xp, wqb, wkb, wvb, bq, bk, bv,
                                            qbuf, kbuf, vtb);

  // 512 blocks: (b,h) x 16 query-groups, 8 waves each
  attn_kernel<<<B_ * H_ * (L_ / 128), 256, 0, stream>>>(qbuf, kbuf, vtb, obuf);

  proj_kernel<<<4096 / 8, 256, 0, stream>>>(obuf, wob, bo, x, out);
}
